// scGNN_28887950033749
// MI455X (gfx1250) — matrix operand order repaired and checked by reading, not compiled, not run on hardware
//
#include <hip/hip_runtime.h>
#include <hip/hip_bf16.h>

// ---------------------------------------------------------------------------
// Types for CDNA5 WMMA (wave32): v_wmma_f32_16x16x32_bf16
// ---------------------------------------------------------------------------
typedef __attribute__((ext_vector_type(16))) __bf16 v16bf;
typedef __attribute__((ext_vector_type(8)))  __bf16 v8bf;
typedef __attribute__((ext_vector_type(8)))  float  v8f;
typedef __attribute__((ext_vector_type(4)))  unsigned short us4;

union AFrag {
    v16bf bf;
    v8bf  h[2];
};

__device__ __forceinline__ unsigned short f2bfu(float f) {
    // round-to-nearest-even float32 -> bf16 bit pattern
    unsigned u = __float_as_uint(f);
    return (unsigned short)((u + 0x7FFFu + ((u >> 16) & 1u)) >> 16);
}

// ---------------------------------------------------------------------------
// Degree / normalization kernels
// ---------------------------------------------------------------------------
__global__ void deg_init_kernel(float* __restrict__ deg, long long n) {
    long long i = (long long)blockIdx.x * blockDim.x + threadIdx.x;
    if (i < n) deg[i] = 1.0f;   // self loop contributes 1
}

__global__ void deg_count_kernel(const long long* __restrict__ dst,
                                 float* __restrict__ deg, long long nE) {
    long long e = (long long)blockIdx.x * blockDim.x + threadIdx.x;
    if (e < nE) atomicAdd(&deg[dst[e]], 1.0f);
}

__global__ void deg_rsqrt_kernel(float* __restrict__ deg, long long n) {
    long long i = (long long)blockIdx.x * blockDim.x + threadIdx.x;
    if (i < n) deg[i] = rsqrtf(deg[i]);   // deg >= 1 always (self loops)
}

// ---------------------------------------------------------------------------
// Streaming fp32 -> bf16 conversion (4 elements / thread)
// ---------------------------------------------------------------------------
__global__ void cvt_bf16_kernel(const float* __restrict__ in,
                                unsigned short* __restrict__ out, long long n4) {
    long long i = (long long)blockIdx.x * blockDim.x + threadIdx.x;
    if (i >= n4) return;
    float4 v = ((const float4*)in)[i];
    us4 o = { f2bfu(v.x), f2bfu(v.y), f2bfu(v.z), f2bfu(v.w) };
    ((us4*)out)[i] = o;
}

// fp32 B[K,N] -> bf16 Bt[N,K] (transpose, write-coalesced); small matrices
__global__ void cvt_bf16_transpose_kernel(const float* __restrict__ B,
                                          unsigned short* __restrict__ Bt,
                                          int K, int N) {
    int idx = blockIdx.x * blockDim.x + threadIdx.x;   // over N*K
    if (idx >= K * N) return;
    int n = idx / K;
    int k = idx - n * K;
    Bt[idx] = f2bfu(B[(size_t)k * N + n]);
}

// ---------------------------------------------------------------------------
// bf16 WMMA GEMM: C[M,N] = A[M,K] * Bt[N,K]^T  (bf16 in, fp32 accumulate).
// One wave computes a 16(M) x 64(N) strip; 4 WMMAs per 32-wide K step,
// A fragment reused in registers, all operand loads contiguous b128s.
// ---------------------------------------------------------------------------
__global__ __launch_bounds__(128)
void gemm_bf16_wmma(const unsigned short* __restrict__ A,   // [M,K] bf16
                    const unsigned short* __restrict__ Bt,  // [N,K] bf16
                    float* __restrict__ C, int M, int K, int N) {
    const int lane  = threadIdx.x & 31;
    const int wave  = threadIdx.x >> 5;
    const int nGroups = N >> 6;                       // 64 columns per wave
    const int mTiles  = (M + 15) >> 4;
    const int gwave = blockIdx.x * (blockDim.x >> 5) + wave;
    if (gwave >= mTiles * nGroups) return;
    const int mTile  = gwave / nGroups;
    const int nGroup = gwave % nGroups;

    const int m    = lane & 15;
    const int half = lane >> 4;
    const int row  = mTile * 16 + m;
    const int rowc = row < M ? row : (M - 1);         // clamped load row
    const int nBase = nGroup * 64;

    const unsigned short* arow = A + (size_t)rowc * K;
    // B^T rows for this lane's 4 output columns
    const unsigned short* bp0 = Bt + (size_t)(nBase +  0 + m) * K;
    const unsigned short* bp1 = Bt + (size_t)(nBase + 16 + m) * K;
    const unsigned short* bp2 = Bt + (size_t)(nBase + 32 + m) * K;
    const unsigned short* bp3 = Bt + (size_t)(nBase + 48 + m) * K;

    v8f acc0 = {}, acc1 = {}, acc2 = {}, acc3 = {};

#pragma unroll 2
    for (int kb = 0; kb < K; kb += 32) {
        // A fragment (16-bit A 16x32), contiguous K order matching B:
        // lane half 0: K = kb+0..15 ; lane half 1: K = kb+16..31
        AFrag a;
        a.h[0] = *(const v8bf*)(arow + kb + half * 16);
        a.h[1] = *(const v8bf*)(arow + kb + half * 16 + 8);

        // B fragments (16-bit B 32x16): lane holds column (nBase+off+m),
        // half 0: K = kb+0..15 ; half 1: K = kb+16..31 -> contiguous in B^T
        const int bo = kb + half * 16;
        v16bf b0 = *(const v16bf*)(bp0 + bo);
        v16bf b1 = *(const v16bf*)(bp1 + bo);
        v16bf b2 = *(const v16bf*)(bp2 + bo);
        v16bf b3 = *(const v16bf*)(bp3 + bo);

        acc0 = __builtin_amdgcn_wmma_f32_16x16x32_bf16(false, a.bf, false, b0,
                                                       (short)0, acc0, false, false);
        acc1 = __builtin_amdgcn_wmma_f32_16x16x32_bf16(false, a.bf, false, b1,
                                                       (short)0, acc1, false, false);
        acc2 = __builtin_amdgcn_wmma_f32_16x16x32_bf16(false, a.bf, false, b2,
                                                       (short)0, acc2, false, false);
        acc3 = __builtin_amdgcn_wmma_f32_16x16x32_bf16(false, a.bf, false, b3,
                                                       (short)0, acc3, false, false);
    }

    // C/D layout: VGPR r -> row mTile*16 + r + 8*half, column = m
#pragma unroll
    for (int r = 0; r < 8; ++r) {
        int orow = mTile * 16 + r + 8 * half;
        if (orow < M) {
            float* c = C + (size_t)orow * N + nBase + m;
            c[0]  = acc0[r];
            c[16] = acc1[r];
            c[32] = acc2[r];
            c[48] = acc3[r];
        }
    }
}

// ---------------------------------------------------------------------------
// Self-loop term (+ optional bias): out[i,f] = feat[i,f]*dinv[i]^2 (+ bias[f])
// Serves as the accumulator initializer before the edge scatter.
// ---------------------------------------------------------------------------
__global__ void selfloop_init_kernel(const float* __restrict__ feat,
                                     const float* __restrict__ dinv,
                                     const float* __restrict__ bias,
                                     float* __restrict__ out,
                                     int F, long long total, int addBias) {
    long long idx = (long long)blockIdx.x * blockDim.x + threadIdx.x;
    if (idx >= total) return;
    long long i = idx / F;
    int f = (int)(idx - i * F);
    float d = dinv[i];
    float v = feat[idx] * d * d;
    if (addBias) v += bias[f];
    out[idx] = v;
}

// ---------------------------------------------------------------------------
// Edge scatter: out[dst] += feat[src] * (dinv[src]*dinv[dst])
// F/4 threads per edge, float4 gathers + per-float global atomics.
// ---------------------------------------------------------------------------
__global__ void scatter_kernel(const float* __restrict__ feat,
                               const long long* __restrict__ sidx,
                               const long long* __restrict__ didx,
                               const float* __restrict__ dinv,
                               float* __restrict__ out,
                               int F, long long nE) {
    const int tpe = F >> 2;                       // threads per edge
    const int epb = blockDim.x / tpe;             // edges per block
    long long e = (long long)blockIdx.x * epb + threadIdx.x / tpe;
    if (e >= nE) return;
    const int f4 = threadIdx.x % tpe;

    long long s = sidx[e];
    long long d = didx[e];
    float nrm = dinv[s] * dinv[d];

    const float4 v = ((const float4*)(feat + (size_t)s * F))[f4];
    float* o = out + (size_t)d * F + (size_t)f4 * 4;
    atomicAdd(o + 0, v.x * nrm);
    atomicAdd(o + 1, v.y * nrm);
    atomicAdd(o + 2, v.z * nrm);
    atomicAdd(o + 3, v.w * nrm);
}

// ---------------------------------------------------------------------------
// Bias + ReLU fused with fp32 -> bf16 conversion (produces GEMM2's A operand)
// ---------------------------------------------------------------------------
__global__ void bias_relu_cvt_kernel(const float* __restrict__ h,
                                     const float* __restrict__ bias,
                                     unsigned short* __restrict__ hbf,
                                     int F, long long total) {
    long long idx = (long long)blockIdx.x * blockDim.x + threadIdx.x;
    if (idx >= total) return;
    int f = (int)(idx % F);
    hbf[idx] = f2bfu(fmaxf(h[idx] + bias[f], 0.0f));
}

// ---------------------------------------------------------------------------
// Host-side launcher
// ---------------------------------------------------------------------------
extern "C" void kernel_launch(void* const* d_in, const int* in_sizes, int n_in,
                              void* d_out, int out_size, void* d_ws, size_t ws_size,
                              hipStream_t stream) {
    const float*      x  = (const float*)d_in[0];
    const long long*  ei = (const long long*)d_in[1];   // int64 edge_index [2, E]
    const float*      W1 = (const float*)d_in[2];
    const float*      b1 = (const float*)d_in[3];
    const float*      W2 = (const float*)d_in[4];
    const float*      b2 = (const float*)d_in[5];

    const int MID = in_sizes[3];                        // 256
    const int IN  = in_sizes[5];                        // 1024
    const long long nNodes = (long long)in_sizes[0] / IN;   // 50000
    const long long nE     = (long long)in_sizes[1] / 2;    // 1.6M
    const long long* src = ei;
    const long long* dst = ei + nE;

    // ---------------- Workspace layout (aliased, ~232 MB total) -----------
    // [deg 256KB][ region R: phaseA = x_bf | xw | h_fp32 ; phaseB = hw ]
    // [h_bf][W1t][W2t]
    char* ws = (char*)d_ws;
    float* deg = (float*)ws;
    size_t off = 256 * 1024;
    char* regionR = ws + off;
    const size_t szXbf = (size_t)nNodes * IN * sizeof(unsigned short);
    const size_t szXw  = (size_t)nNodes * MID * sizeof(float);
    const size_t szHw  = (size_t)nNodes * IN * sizeof(float);
    unsigned short* xbf = (unsigned short*)regionR;
    float* xw = (float*)(regionR + szXbf);
    float* h  = (float*)(regionR + szXbf + szXw);
    float* hw = (float*)regionR;                         // aliases xbf|xw|h
    size_t szR = szXbf + 2 * szXw;
    if (szHw > szR) szR = szHw;
    off += szR;
    unsigned short* hbf = (unsigned short*)(ws + off);
    off += (size_t)nNodes * MID * sizeof(unsigned short);
    unsigned short* W1t = (unsigned short*)(ws + off);
    off += (size_t)IN * MID * sizeof(unsigned short);
    unsigned short* W2t = (unsigned short*)(ws + off);
    float* out = (float*)d_out;

    const int T = 256;
    // 1) degrees with self-loops -> dinv = rsqrt(deg)
    deg_init_kernel<<<(unsigned)((nNodes + T - 1) / T), T, 0, stream>>>(deg, nNodes);
    deg_count_kernel<<<(unsigned)((nE + T - 1) / T), T, 0, stream>>>(dst, deg, nE);
    deg_rsqrt_kernel<<<(unsigned)((nNodes + T - 1) / T), T, 0, stream>>>(deg, nNodes);

    // 2) operand conversions: x -> bf16, W1/W2 -> bf16 transposed
    {
        long long n4 = nNodes * IN / 4;
        cvt_bf16_kernel<<<(unsigned)((n4 + T - 1) / T), T, 0, stream>>>(x, xbf, n4);
        int kn = IN * MID;
        cvt_bf16_transpose_kernel<<<(kn + T - 1) / T, T, 0, stream>>>(W1, W1t, IN, MID);
        cvt_bf16_transpose_kernel<<<(kn + T - 1) / T, T, 0, stream>>>(W2, W2t, MID, IN);
    }

    // 3) xw = x @ W1  (bf16 WMMA, fp32 accumulate)
    {
        int mTiles = (int)((nNodes + 15) / 16);
        int waves  = mTiles * (MID >> 6);
        gemm_bf16_wmma<<<(waves + 3) / 4, 128, 0, stream>>>(xbf, W1t, xw,
                                                            (int)nNodes, IN, MID);
    }

    // 4) h = selfloop(xw) ; scatter edges ; bias+relu -> bf16 (GEMM2 operand)
    {
        long long total = nNodes * MID;
        selfloop_init_kernel<<<(unsigned)((total + T - 1) / T), T, 0, stream>>>(
            xw, deg, b1, h, MID, total, 0);
        int epb = T / (MID >> 2);
        scatter_kernel<<<(unsigned)((nE + epb - 1) / epb), T, 0, stream>>>(
            xw, src, dst, deg, h, MID, nE);
        bias_relu_cvt_kernel<<<(unsigned)((total + T - 1) / T), T, 0, stream>>>(
            h, b1, hbf, MID, total);
    }

    // 5) hw = h @ W2  (hw aliases the now-dead xbf/xw/h region)
    {
        int mTiles = (int)((nNodes + 15) / 16);
        int waves  = mTiles * (IN >> 6);
        gemm_bf16_wmma<<<(waves + 3) / 4, 128, 0, stream>>>(hbf, W2t, hw,
                                                            (int)nNodes, MID, IN);
    }

    // 6) out = selfloop(hw) + b2 ; scatter edges into d_out
    {
        long long total = nNodes * IN;
        selfloop_init_kernel<<<(unsigned)((total + T - 1) / T), T, 0, stream>>>(
            hw, deg, b2, out, IN, total, 1);
        int epb = T / (IN >> 2);   // == 1
        scatter_kernel<<<(unsigned)((nE + epb - 1) / epb), T, 0, stream>>>(
            hw, src, dst, deg, out, IN, nE);
    }
}
